// GPT_60662118088964
// MI455X (gfx1250) — matrix operand order repaired and checked by reading, not compiled, not run on hardware
//
#include <hip/hip_runtime.h>
#include <cstdint>
#include <cstddef>

#define B_  4
#define T_  1024
#define C_  768
#define L_  12
#define H_  12
#define V_  50304
#define D_  64
#define BT_ (B_ * T_)
#define C3_ (3 * C_)
#define C4_ (4 * C_)

typedef __bf16 bf16_t;
typedef bf16_t v16bf __attribute__((ext_vector_type(16)));
typedef float  v8f   __attribute__((ext_vector_type(8)));

union FragBF {
    v16bf v;
    uint4 q[2];
};

__device__ inline unsigned short f2bf(float x) {
    unsigned int u = __float_as_uint(x);
    unsigned int r = u + 0x7FFFu + ((u >> 16) & 1u);
    return (unsigned short)(r >> 16);
}

__device__ inline v8f vzero8() {
    v8f z;
#pragma unroll
    for (int i = 0; i < 8; ++i) z[i] = 0.0f;
    return z;
}

// ---------------------------------------------------------------------------
// Async global -> LDS copy (16B per lane), CDNA5 path (ASYNCcnt).
// Builtin signature (probed): (v4i addrspace(1)*, v4i addrspace(3)*, Ii, Ii)
// ---------------------------------------------------------------------------
__device__ inline void async_copy_b128(const unsigned short* gsrc, unsigned short* ldst) {
#if __has_builtin(__builtin_amdgcn_global_load_async_to_lds_b128)
    typedef int v4i_g __attribute__((vector_size(16)));
    typedef __attribute__((address_space(1))) v4i_g* gp_t;
    typedef __attribute__((address_space(3))) v4i_g* lp_t;
    __builtin_amdgcn_global_load_async_to_lds_b128(
        (gp_t)(void*)const_cast<unsigned short*>(gsrc),
        (lp_t)(void*)ldst, 0, 0);
#else
    unsigned int lds = (unsigned int)(size_t)ldst;           // low 32b = LDS offset
    unsigned long long ga = (unsigned long long)(size_t)gsrc;
    asm volatile("global_load_async_to_lds_b128 %0, %1, off"
                 :: "v"(lds), "v"(ga) : "memory");
#endif
}

__device__ inline void wait_async(int n) {
#if __has_builtin(__builtin_amdgcn_s_wait_asynccnt)
    if (n == 0) __builtin_amdgcn_s_wait_asynccnt(0);
    else        __builtin_amdgcn_s_wait_asynccnt(1);
#else
    if (n == 0) asm volatile("s_wait_asynccnt 0x0" ::: "memory");
    else        asm volatile("s_wait_asynccnt 0x1" ::: "memory");
#endif
}

// ---------------------------------------------------------------------------
// WMMA fragment loaders (wave32, 16-bit data).
// A (16x32, MxK): lanes 0-15 m=lane, K 0..7 & 16..23; lanes 16-31 K 8..15 & 24..31.
// B (32x16, KxN): lanes 0-15 n=lane, K 0..7 & 16..23; lanes 16-31 K 8..15 & 24..31.
// A source row-major [M,K]; B source row-major B^T, i.e. [N,K].
// ---------------------------------------------------------------------------
__device__ inline v16bf load_frag_a(const unsigned short* A, int lda) {
    int lane = threadIdx.x & 31;
    int m  = lane & 15;
    int kb = (lane >> 4) << 3;
    const unsigned short* p = A + (size_t)m * lda + kb;
    FragBF f;
    f.q[0] = *(const uint4*)(p);
    f.q[1] = *(const uint4*)(p + 16);
    return f.v;
}

__device__ inline v16bf load_frag_b(const unsigned short* BT, int ldb) {
    int lane = threadIdx.x & 31;
    int n  = lane & 15;
    int kb = (lane >> 4) << 3;
    const unsigned short* p = BT + (size_t)n * ldb + kb;
    FragBF f;
    f.q[0] = *(const uint4*)(p);
    f.q[1] = *(const uint4*)(p + 16);
    return f.v;
}

__device__ inline v8f wmma_bf16(v16bf a, v16bf b, v8f c) {
    return __builtin_amdgcn_wmma_f32_16x16x32_bf16(
        false, a, false, b, (short)0, c, false, false);
}

// ---------------------------------------------------------------------------
// GEMM: out[M,N] = A[M,K](bf16) * BT[N,K]^T(bf16) + epilogue.
// 256 threads = 8 waves per block; each wave computes a 32x64 tile
// (block = 256 rows x 64 cols). B tile (64x32) is double-buffered in LDS via
// async global->LDS loads shared by all 8 waves.
// ---------------------------------------------------------------------------
enum { EPI_QKV = 0, EPI_BF16_BIAS_GELU = 1, EPI_F32_BIAS_RES = 2, EPI_F32_PLAIN = 3 };

template <int EPI>
__global__ __launch_bounds__(256) void gemm_bf16_kernel(
    const unsigned short* __restrict__ A, const unsigned short* __restrict__ BT,
    const float* __restrict__ bias, const float* __restrict__ resid,
    void* __restrict__ outv, unsigned short* __restrict__ vTout,
    int K, int lda, int ldb, int ldo)
{
    __shared__ __align__(16) unsigned short Bl[2][64 * 32];

    const int tid  = threadIdx.x;
    const int lane = tid & 31;
    const int wave = tid >> 5;
    const int row0 = (blockIdx.y * 8 + wave) << 5;   // 32 rows per wave
    const int col0 = blockIdx.x << 6;

    // async-stage mapping: each of the 256 threads copies 16B of the B tile
    const int sn = tid >> 2;            // 0..63 column of the 64-wide tile
    const int sk = (tid & 3) << 3;      // 0,8,16,24 within the 32-k slab
    const unsigned short* bsrc = BT + (size_t)(col0 + sn) * ldb + sk;

    v8f acc[2][4];
#pragma unroll
    for (int s = 0; s < 2; ++s)
#pragma unroll
        for (int j = 0; j < 4; ++j) acc[s][j] = vzero8();

    const unsigned short* Ar0 = A + (size_t)row0 * lda;
    const unsigned short* Ar1 = A + (size_t)(row0 + 16) * lda;

    const int niter = K >> 5;
    async_copy_b128(bsrc, &Bl[0][sn * 32 + sk]);
    for (int i = 0; i < niter; ++i) {
        const int k0 = i << 5;
        if (i + 1 < niter) {
            async_copy_b128(bsrc + (k0 + 32), &Bl[(i + 1) & 1][sn * 32 + sk]);
            wait_async(1);
        } else {
            wait_async(0);
        }
        __syncthreads();
        v16bf a0 = load_frag_a(Ar0 + k0, lda);
        v16bf a1 = load_frag_a(Ar1 + k0, lda);
        const unsigned short* bb = &Bl[i & 1][0];
#pragma unroll
        for (int j = 0; j < 4; ++j) {
            v16bf b = load_frag_b(bb + (size_t)(16 * j) * 32, 32);
            acc[0][j] = wmma_bf16(a0, b, acc[0][j]);
            acc[1][j] = wmma_bf16(a1, b, acc[1][j]);
        }
        __syncthreads();
    }

    const int n0 = lane & 15;
    const int hi = (lane >> 4) << 3;
#pragma unroll
    for (int s = 0; s < 2; ++s) {
        const int rowb = row0 + 16 * s;
#pragma unroll
        for (int j = 0; j < 4; ++j) {
            const int col = col0 + 16 * j + n0;
            float bv = (EPI == EPI_F32_PLAIN) ? 0.0f : bias[col];
#pragma unroll
            for (int r = 0; r < 8; ++r) {
                const int row = rowb + hi + r;
                float v = acc[s][j][r] + bv;
                const size_t o = (size_t)row * ldo + col;
                if (EPI == EPI_BF16_BIAS_GELU) {
                    v = 0.5f * v * (1.0f + erff(v * 0.70710678118654752f));
                    ((unsigned short*)outv)[o] = f2bf(v);
                } else if (EPI == EPI_QKV) {
                    if (col0 + 16 * j < 2 * C_) {
                        ((unsigned short*)outv)[o] = f2bf(v);   // q,k in [BT,3C]
                    } else {
                        const int b = row >> 10;                // row / T_
                        const int t = row & (T_ - 1);
                        vTout[((size_t)b * C_ + (col - 2 * C_)) * T_ + t] = f2bf(v);
                    }
                } else if (EPI == EPI_F32_BIAS_RES) {
                    ((float*)outv)[o] = resid[o] + v;
                } else {
                    ((float*)outv)[o] = v;
                }
            }
        }
    }
}

// ---------------------------------------------------------------------------
// Flash attention: one wave per (b, h, 16-query tile). Causal, scale=1/8.
// qkv: bf16 [B*T, 3C] (q,k thirds); vT: bf16 [B, C, T]; y: bf16 [B*T, C].
// ---------------------------------------------------------------------------
template <bool MASKED>
__device__ __forceinline__ void attn_step(
    int kt, int q0, int n0, int hi, int ld,
    v16bf qa0, v16bf qa1,
    const unsigned short* kbase, const unsigned short* vtb,
    float rm[8], float rl[8], v8f o[4], unsigned short* Pl)
{
    const float scale = 0.125f;
    v8f S0 = vzero8(), S1 = vzero8();
    S0 = wmma_bf16(qa0, load_frag_b(kbase + (size_t)kt * ld, ld), S0);
    S0 = wmma_bf16(qa1, load_frag_b(kbase + (size_t)kt * ld + 32, ld), S0);
    S1 = wmma_bf16(qa0, load_frag_b(kbase + (size_t)(kt + 16) * ld, ld), S1);
    S1 = wmma_bf16(qa1, load_frag_b(kbase + (size_t)(kt + 16) * ld + 32, ld), S1);

#pragma unroll
    for (int r = 0; r < 8; ++r) {
        float s0 = S0[r] * scale;
        float s1 = S1[r] * scale;
        if (MASKED) {
            const int row = q0 + hi + r;
            if (kt + n0      > row) s0 = -1e30f;
            if (kt + 16 + n0 > row) s1 = -1e30f;
        }
        float mx = fmaxf(s0, s1);
        for (int m = 1; m <= 8; m <<= 1) mx = fmaxf(mx, __shfl_xor(mx, m, 32));
        const float mnew = fmaxf(rm[r], mx);
        const float e0 = __expf(s0 - mnew);
        const float e1 = __expf(s1 - mnew);
        float rs = e0 + e1;
        for (int m = 1; m <= 8; m <<= 1) rs += __shfl_xor(rs, m, 32);
        const float corr = __expf(rm[r] - mnew);
        rl[r] = rl[r] * corr + rs;
        rm[r] = mnew;
#pragma unroll
        for (int j = 0; j < 4; ++j) o[j][r] *= corr;
        Pl[(hi + r) * 32 + n0]      = f2bf(e0);
        Pl[(hi + r) * 32 + 16 + n0] = f2bf(e1);
    }
    __syncthreads();
    const v16bf pa = load_frag_a(Pl, 32);
#pragma unroll
    for (int j = 0; j < 4; ++j)
        o[j] = wmma_bf16(pa, load_frag_b(vtb + (size_t)(16 * j) * T_ + kt, T_), o[j]);
    __syncthreads();
}

__global__ __launch_bounds__(32) void attn_kernel(
    const unsigned short* __restrict__ qkv,
    const unsigned short* __restrict__ vT,
    unsigned short* __restrict__ y)
{
    const int lane = threadIdx.x & 31;
    const int q0 = blockIdx.x << 4;
    const int bh = blockIdx.y;
    const int b = bh / H_, h = bh % H_;
    const int ld = C3_;

    const unsigned short* qbase = qkv + (size_t)b * T_ * ld + h * D_;
    const unsigned short* kbase = qkv + (size_t)b * T_ * ld + C_ + h * D_;
    const unsigned short* vtb   = vT + ((size_t)b * C_ + h * D_) * T_;

    const v16bf qa0 = load_frag_a(qbase + (size_t)q0 * ld, ld);
    const v16bf qa1 = load_frag_a(qbase + (size_t)q0 * ld + 32, ld);

    float rm[8], rl[8];
    v8f o[4];
#pragma unroll
    for (int r = 0; r < 8; ++r) { rm[r] = -1e30f; rl[r] = 0.0f; }
#pragma unroll
    for (int j = 0; j < 4; ++j) o[j] = vzero8();

    __shared__ __align__(16) unsigned short Pl[16 * 32];

    const int n0 = lane & 15;
    const int hi = (lane >> 4) << 3;

    // key blocks entirely below the diagonal need no mask
    const int kfull_end = (q0 >= 31) ? ((((q0 - 31) >> 5) + 1) << 5) : 0;
    int kt = 0;
    for (; kt < kfull_end; kt += 32)
        attn_step<false>(kt, q0, n0, hi, ld, qa0, qa1, kbase, vtb, rm, rl, o, Pl);
    for (; kt < q0 + 16; kt += 32)
        attn_step<true>(kt, q0, n0, hi, ld, qa0, qa1, kbase, vtb, rm, rl, o, Pl);

    unsigned short* yb = y + (size_t)(b * T_ + q0) * C_ + h * D_;
#pragma unroll
    for (int r = 0; r < 8; ++r) {
        const float inv = 1.0f / rl[r];
#pragma unroll
        for (int j = 0; j < 4; ++j)
            yb[(size_t)(hi + r) * C_ + 16 * j + n0] = f2bf(o[j][r] * inv);
    }
}

// ---------------------------------------------------------------------------
// Helper kernels
// ---------------------------------------------------------------------------
__global__ void embed_kernel(const int* __restrict__ idx, const float* __restrict__ wte,
                             const float* __restrict__ wpe, float* __restrict__ x)
{
    size_t i = (size_t)blockIdx.x * blockDim.x + threadIdx.x;
    if (i >= (size_t)BT_ * C_) return;
    int c = (int)(i % C_);
    size_t bt = i / C_;
    int t = (int)(bt % T_);
    x[i] = wte[(size_t)idx[bt] * C_ + c] + wpe[(size_t)t * C_ + c];
}

__global__ __launch_bounds__(32) void ln_kernel(const float* __restrict__ x,
    const float* __restrict__ w, const float* __restrict__ bp,
    unsigned short* __restrict__ out)
{
    const int row = blockIdx.x;
    const int lane = threadIdx.x;
    const float* xr = x + (size_t)row * C_;
    float s = 0.0f, s2 = 0.0f;
    for (int c = lane; c < C_; c += 32) { float v = xr[c]; s += v; s2 += v * v; }
    for (int m = 1; m < 32; m <<= 1) { s += __shfl_xor(s, m, 32); s2 += __shfl_xor(s2, m, 32); }
    const float mean = s * (1.0f / C_);
    const float var  = s2 * (1.0f / C_) - mean * mean;
    const float inv  = rsqrtf(var + 1e-5f);
    unsigned short* orow = out + (size_t)row * C_;
    for (int c = lane; c < C_; c += 32)
        orow[c] = f2bf((xr[c] - mean) * inv * w[c] + bp[c]);
}

// in: f32 [L][K,N] -> out: bf16 [L][N,K]
__global__ void transpose_bf16_kernel(const float* __restrict__ in,
                                      unsigned short* __restrict__ out, int Kd, int Nd)
{
    const int l = blockIdx.y;
    const float* src = in + (size_t)l * Kd * Nd;
    unsigned short* dst = out + (size_t)l * Kd * Nd;
    const size_t n_el = (size_t)Kd * Nd;
    for (size_t i = (size_t)blockIdx.x * blockDim.x + threadIdx.x; i < n_el;
         i += (size_t)gridDim.x * blockDim.x) {
        int n = (int)(i % Nd);
        int k = (int)(i / Nd);
        dst[(size_t)n * Kd + k] = f2bf(src[i]);
    }
}

__global__ void copy_bf16_kernel(const float* __restrict__ in,
                                 unsigned short* __restrict__ out, size_t n_el)
{
    for (size_t i = (size_t)blockIdx.x * blockDim.x + threadIdx.x; i < n_el;
         i += (size_t)gridDim.x * blockDim.x)
        out[i] = f2bf(in[i]);
}

__global__ __launch_bounds__(256) void loss_kernel(const float* __restrict__ logits,
    const int* __restrict__ targets, float* __restrict__ accum)
{
    const int row = blockIdx.x;
    const float* lr = logits + (size_t)row * V_;
    __shared__ float red[256];
    const int tid = threadIdx.x;
    float mx = -1e30f;
    for (int v = tid; v < V_; v += 256) mx = fmaxf(mx, lr[v]);
    red[tid] = mx; __syncthreads();
    for (int s = 128; s > 0; s >>= 1) {
        if (tid < s) red[tid] = fmaxf(red[tid], red[tid + s]);
        __syncthreads();
    }
    mx = red[0]; __syncthreads();
    float se = 0.0f;
    for (int v = tid; v < V_; v += 256) se += __expf(lr[v] - mx);
    red[tid] = se; __syncthreads();
    for (int s = 128; s > 0; s >>= 1) {
        if (tid < s) red[tid] += red[tid + s];
        __syncthreads();
    }
    se = red[0];
    if (tid == 0) {
        const int tgt = targets[row];
        if (tgt != -1) {
            const float nll = -(lr[tgt] - mx - logf(se));
            atomicAdd(&accum[0], nll);
            atomicAdd(&accum[1], 1.0f);
        }
    }
}

__global__ void zero2_kernel(float* a) { if (threadIdx.x < 2) a[threadIdx.x] = 0.0f; }
__global__ void finalize_kernel(const float* a, float* out) { out[0] = a[0] / a[1]; }

// ---------------------------------------------------------------------------
extern "C" void kernel_launch(void* const* d_in, const int* in_sizes, int n_in,
                              void* d_out, int out_size, void* d_ws, size_t ws_size,
                              hipStream_t stream) {
    (void)in_sizes; (void)n_in; (void)out_size; (void)ws_size;

    const int*   idx      = (const int*)  d_in[0];
    const int*   targets  = (const int*)  d_in[1];
    const float* wte      = (const float*)d_in[2];
    const float* wpe      = (const float*)d_in[3];
    const float* ln1_w    = (const float*)d_in[4];
    const float* ln1_b    = (const float*)d_in[5];
    const float* attn_w   = (const float*)d_in[6];
    const float* attn_b   = (const float*)d_in[7];
    const float* attn_pw  = (const float*)d_in[8];
    const float* attn_pb  = (const float*)d_in[9];
    const float* ln2_w    = (const float*)d_in[10];
    const float* ln2_b    = (const float*)d_in[11];
    const float* fc_w     = (const float*)d_in[12];
    const float* fc_b     = (const float*)d_in[13];
    const float* mlp_pw   = (const float*)d_in[14];
    const float* mlp_pb   = (const float*)d_in[15];
    const float* lnf_w    = (const float*)d_in[16];
    const float* lnf_b    = (const float*)d_in[17];

    char* wp = (char*)d_ws;
    auto carve = [&](size_t bytes) -> void* {
        void* p = (void*)wp;
        wp += (bytes + 255) & ~(size_t)255;
        return p;
    };
    float*          x    = (float*)         carve((size_t)BT_ * C_ * 4);
    unsigned short* h    = (unsigned short*)carve((size_t)BT_ * C_ * 2);
    unsigned short* qkv  = (unsigned short*)carve((size_t)BT_ * C3_ * 2);
    unsigned short* vT   = (unsigned short*)carve((size_t)B_ * C_ * T_ * 2);
    unsigned short* yb   = (unsigned short*)carve((size_t)BT_ * C_ * 2);
    unsigned short* act  = (unsigned short*)carve((size_t)BT_ * C4_ * 2);
    unsigned short* awT  = (unsigned short*)carve((size_t)L_ * C_ * C3_ * 2);
    unsigned short* pwT  = (unsigned short*)carve((size_t)L_ * C_ * C_ * 2);
    unsigned short* fwT  = (unsigned short*)carve((size_t)L_ * C_ * C4_ * 2);
    unsigned short* mwT  = (unsigned short*)carve((size_t)L_ * C4_ * C_ * 2);
    unsigned short* wteb = (unsigned short*)carve((size_t)V_ * C_ * 2);
    float*          accum = (float*)        carve(2 * 4);

    // one-time (per launch) weight conversion
    transpose_bf16_kernel<<<dim3(1024, L_), 256, 0, stream>>>(attn_w,  awT, C_,  C3_);
    transpose_bf16_kernel<<<dim3(1024, L_), 256, 0, stream>>>(attn_pw, pwT, C_,  C_);
    transpose_bf16_kernel<<<dim3(1024, L_), 256, 0, stream>>>(fc_w,    fwT, C_,  C4_);
    transpose_bf16_kernel<<<dim3(1024, L_), 256, 0, stream>>>(mlp_pw,  mwT, C4_, C_);
    copy_bf16_kernel<<<4096, 256, 0, stream>>>(wte, wteb, (size_t)V_ * C_);

    // embedding
    embed_kernel<<<((size_t)BT_ * C_ + 255) / 256, 256, 0, stream>>>(idx, wte, wpe, x);

    // transformer blocks
    for (int l = 0; l < L_; ++l) {
        ln_kernel<<<BT_, 32, 0, stream>>>(x, ln1_w + (size_t)l * C_, ln1_b + (size_t)l * C_, h);

        gemm_bf16_kernel<EPI_QKV><<<dim3(C3_ / 64, BT_ / 256), 256, 0, stream>>>(
            h, awT + (size_t)l * C_ * C3_, attn_b + (size_t)l * C3_, nullptr,
            (void*)qkv, vT, C_, C_, C_, C3_);

        attn_kernel<<<dim3(T_ / 16, B_ * H_), 32, 0, stream>>>(qkv, vT, yb);

        gemm_bf16_kernel<EPI_F32_BIAS_RES><<<dim3(C_ / 64, BT_ / 256), 256, 0, stream>>>(
            yb, pwT + (size_t)l * C_ * C_, attn_pb + (size_t)l * C_, x,
            (void*)x, nullptr, C_, C_, C_, C_);

        ln_kernel<<<BT_, 32, 0, stream>>>(x, ln2_w + (size_t)l * C_, ln2_b + (size_t)l * C_, h);

        gemm_bf16_kernel<EPI_BF16_BIAS_GELU><<<dim3(C4_ / 64, BT_ / 256), 256, 0, stream>>>(
            h, fwT + (size_t)l * C_ * C4_, fc_b + (size_t)l * C4_, nullptr,
            (void*)act, nullptr, C_, C_, C_, C4_);

        gemm_bf16_kernel<EPI_F32_BIAS_RES><<<dim3(C_ / 64, BT_ / 256), 256, 0, stream>>>(
            act, mwT + (size_t)l * C4_ * C_, mlp_pb + (size_t)l * C_, x,
            (void*)x, nullptr, C4_, C4_, C4_, C_);
    }

    // final LN + tied lm_head
    ln_kernel<<<BT_, 32, 0, stream>>>(x, lnf_w, lnf_b, h);

    float* logits = (float*)d_out;
    gemm_bf16_kernel<EPI_F32_PLAIN><<<dim3(V_ / 64, BT_ / 256), 256, 0, stream>>>(
        h, wteb, nullptr, nullptr, (void*)logits, nullptr, C_, C_, C_, V_);

    // loss
    zero2_kernel<<<1, 32, 0, stream>>>(accum);
    loss_kernel<<<BT_, 256, 0, stream>>>(logits, targets, accum);
    finalize_kernel<<<1, 1, 0, stream>>>(accum, logits + (size_t)BT_ * V_);
}
